// LGTM_36782099923056
// MI455X (gfx1250) — matrix operand order repaired and checked by reading, compiled, not run on hardware
//
#include <hip/hip_runtime.h>
#include <hip/hip_bf16.h>

// ---------------- problem constants (from reference) ----------------
#define Bz   32      // batch
#define Sq   1024    // audio seq len
#define Tt   64      // text seq len
#define Hd   768     // hidden
#define NHd  12      // heads
#define DHd  64      // head dim
#define NLt  32      // latents
#define Llay 2       // layers per block
#define Ktop 64      // top-k

typedef __attribute__((ext_vector_type(16))) __bf16 v16bf;
typedef __attribute__((ext_vector_type(8)))  float  v8f;

union AFrag { v16bf v; unsigned d[8]; };

__device__ __forceinline__ unsigned f2bf1(float x) {
  unsigned u = __float_as_uint(x);
  u += 0x7FFFu + ((u >> 16) & 1u);     // round-to-nearest-even
  return u >> 16;
}
__device__ __forceinline__ unsigned packbf(float a, float b) {
  return f2bf1(a) | (f2bf1(b) << 16);
}

// =====================================================================
// Batched WMMA GEMM:  C[z] = alpha * A[z] x B[z] + bias
//   A: M x Kd (row major, lda), B: Kd x N (ldb) or N x Kd if TRANS_B
//   batch z decomposes as (bb = z/nh, hh = z%nh) with per-operand strides.
// f32 operands converted to bf16 while staging through LDS; fp32 accum
// via v_wmma_f32_16x16x32_bf16. 128x128 tile, 8 wave32 waves (4Mx2N),
// each wave: 32x64 = 2x4 WMMA subtiles.
// =====================================================================
template<bool TRANS_B>
__global__ __launch_bounds__(256) void wmma_gemm(
    const float* __restrict__ A, const float* __restrict__ Bm,
    const float* __restrict__ bias, float* __restrict__ C,
    int M, int N, int Kd, int lda, int ldb, int ldc,
    int nh, long aO, long aI, long bO, long bI, long cO, long cI,
    float alpha)
{
  __shared__ unsigned short As[128][40];   // [m][k], bf16, +8 pad
  __shared__ unsigned short Bs[128][40];   // [n][k], bf16, +8 pad

  const int z  = blockIdx.z;
  const int bb = z / nh, hh = z % nh;
  A  += (long)bb * aO + (long)hh * aI;
  Bm += (long)bb * bO + (long)hh * bI;
  C  += (long)bb * cO + (long)hh * cI;

  const int tid  = threadIdx.x;
  const int lane = tid & 31;
  const int wid  = tid >> 5;
  const int lo   = lane & 15, hi = lane >> 4;
  const int wm   = (wid & 3) << 5;           // wave M offset: 0/32/64/96
  const int wn   = (wid >> 2) << 6;          // wave N offset: 0/64
  const int tM   = blockIdx.y * 128, tN = blockIdx.x * 128;

  v8f acc[2][4];
  #pragma unroll
  for (int i = 0; i < 2; ++i)
    #pragma unroll
    for (int j = 0; j < 4; ++j)
      acc[i][j] = (v8f){0.f,0.f,0.f,0.f,0.f,0.f,0.f,0.f};

  for (int kc = 0; kc < Kd; kc += 32) {
    // ---- stage A tile 128x32 (coalesced float4, convert to bf16) ----
    #pragma unroll
    for (int i = 0; i < 4; ++i) {
      int idx = tid + i * 256;               // 0..1023
      int r = idx >> 3, c4 = (idx & 7) << 2;
      float4 f = {0.f, 0.f, 0.f, 0.f};
      int gr = tM + r;
      if (gr < M) f = *(const float4*)(A + (long)gr * lda + kc + c4);
      *(unsigned*)&As[r][c4]     = packbf(f.x, f.y);
      *(unsigned*)&As[r][c4 + 2] = packbf(f.z, f.w);
    }
    // ---- stage B tile as [n][k] so K-pairs are dword-contiguous ----
    if (TRANS_B) {
      #pragma unroll
      for (int i = 0; i < 4; ++i) {
        int idx = tid + i * 256;
        int n = idx >> 3, k4 = (idx & 7) << 2;
        float4 f = {0.f, 0.f, 0.f, 0.f};
        int gn = tN + n;
        if (gn < N) f = *(const float4*)(Bm + (long)gn * ldb + kc + k4);
        *(unsigned*)&Bs[n][k4]     = packbf(f.x, f.y);
        *(unsigned*)&Bs[n][k4 + 2] = packbf(f.z, f.w);
      }
    } else {
      #pragma unroll
      for (int i = 0; i < 16; ++i) {
        int idx = tid + i * 256;             // 0..4095, coalesced over n
        int k = idx >> 7, n = idx & 127;
        int gn = tN + n;
        float v = 0.f;
        if (gn < N) v = Bm[(long)(kc + k) * ldb + gn];
        Bs[n][k] = (unsigned short)f2bf1(v);
      }
    }
    __syncthreads();

    if (kc + 32 < Kd) {
      // speculative L2 prefetch of the next K tile (global_prefetch_b8)
      __builtin_prefetch(A + (long)(tM + (tid & 127)) * lda + kc + 32, 0, 1);
    }

    // ---- build fragments per ISA bf16 VGPR striping ----
    // A 16x32: lane = M%16, pair p: K = (p<4 ? 2p : 16+2(p-4)) + 8*(lane>=16)
    // B 32x16: lane = N%16, pair p: K = (p<4 ? 2p : 16+2(p-4)) + 8*(lane>=16)
    AFrag a[2], b[4];
    #pragma unroll
    for (int i = 0; i < 2; ++i) {
      int row = wm + (i << 4) + lo;
      #pragma unroll
      for (int p = 0; p < 8; ++p) {
        int kk = ((p < 4) ? (p << 1) : (16 + ((p - 4) << 1))) + (hi << 3);
        a[i].d[p] = *(const unsigned*)&As[row][kk];
      }
    }
    #pragma unroll
    for (int j = 0; j < 4; ++j) {
      int col = wn + (j << 4) + lo;
      #pragma unroll
      for (int p = 0; p < 8; ++p) {
        int kk = ((p < 4) ? (p << 1) : (16 + ((p - 4) << 1))) + (hi << 3);
        b[j].d[p] = *(const unsigned*)&Bs[col][kk];
      }
    }
    #pragma unroll
    for (int i = 0; i < 2; ++i)
      #pragma unroll
      for (int j = 0; j < 4; ++j)
        acc[i][j] = __builtin_amdgcn_wmma_f32_16x16x32_bf16(
            false, a[i].v, false, b[j].v, (short)0, acc[i][j], false, false);
    __syncthreads();
  }

  // ---- store: C/D layout row = r + 8*(lane>=16), col = lane%16 ----
  #pragma unroll
  for (int j = 0; j < 4; ++j) {
    int col = tN + wn + (j << 4) + lo;
    if (col >= N) continue;
    float bv = bias ? bias[col] : 0.f;
    #pragma unroll
    for (int i = 0; i < 2; ++i)
      #pragma unroll
      for (int r = 0; r < 8; ++r) {
        int row = tM + wm + (i << 4) + (hi << 3) + r;
        if (row < M) C[(long)row * ldc + col] = alpha * acc[i][j][r] + bv;
      }
  }
}

// ---------------- softmax over rows (one wave32 per row) ----------------
__global__ __launch_bounds__(256) void softmax_rows(
    float* __restrict__ S, const int* __restrict__ mask,
    int rows, int cols, int nhLq, int maskT)
{
  int r = blockIdx.x * 8 + (threadIdx.x >> 5);
  if (r >= rows) return;
  int lane = threadIdx.x & 31;
  float* row = S + (long)r * cols;
  float buf[32];                      // cols <= 1024
  int n = 0;
  float mx = -3.4e38f;
  if (mask) {
    int bb = r / nhLq;
    for (int c = lane; c < cols; c += 32, ++n) {
      float v = row[c] + (1.f - (float)mask[bb * maskT + c]) * -1e4f;
      buf[n] = v; mx = fmaxf(mx, v);
    }
  } else {
    for (int c = lane; c < cols; c += 32, ++n) {
      float v = row[c]; buf[n] = v; mx = fmaxf(mx, v);
    }
  }
  #pragma unroll
  for (int o = 16; o > 0; o >>= 1) mx = fmaxf(mx, __shfl_xor(mx, o, 32));
  float sum = 0.f;
  for (int i = 0; i < n; ++i) { buf[i] = __expf(buf[i] - mx); sum += buf[i]; }
  #pragma unroll
  for (int o = 16; o > 0; o >>= 1) sum += __shfl_xor(sum, o, 32);
  float inv = 1.f / sum;
  n = 0;
  for (int c = lane; c < cols; c += 32, ++n) row[c] = buf[n] * inv;
}

// ---------------- LayerNorm(x + a) (one wave32 per row, H=768) ----------------
__global__ __launch_bounds__(256) void ln_residual(
    const float* __restrict__ x, const float* __restrict__ a,
    const float* __restrict__ g, const float* __restrict__ b,
    float* __restrict__ out, int rows)
{
  int r = blockIdx.x * 8 + (threadIdx.x >> 5);
  if (r >= rows) return;
  int lane = threadIdx.x & 31;
  const float* xr = x + (long)r * Hd;
  const float* ar = a + (long)r * Hd;
  float v[24];
  float s = 0.f;
  #pragma unroll
  for (int i = 0; i < 24; ++i) { v[i] = xr[lane + i * 32] + ar[lane + i * 32]; s += v[i]; }
  #pragma unroll
  for (int o = 16; o > 0; o >>= 1) s += __shfl_xor(s, o, 32);
  float m = s * (1.f / Hd);
  float q = 0.f;
  #pragma unroll
  for (int i = 0; i < 24; ++i) { float d = v[i] - m; q += d * d; }
  #pragma unroll
  for (int o = 16; o > 0; o >>= 1) q += __shfl_xor(q, o, 32);
  float inv = rsqrtf(q * (1.f / Hd) + 1e-12f);
  float* orow = out + (long)r * Hd;
  #pragma unroll
  for (int i = 0; i < 24; ++i) {
    int c = lane + i * 32;
    orow[c] = (v[i] - m) * inv * g[c] + b[c];
  }
}

// ---------------- small elementwise helpers ----------------
__global__ void bcast_qt(const float* __restrict__ src, float* __restrict__ dst, long n) {
  long i = (long)blockIdx.x * 256 + threadIdx.x;
  if (i < n) dst[i] = src[i % ((long)NLt * Hd)];
}
__global__ void fill_zero(float* __restrict__ p, long n) {
  long i = (long)blockIdx.x * 256 + threadIdx.x;
  if (i < n) p[i] = 0.f;
}
__global__ void copyf(const float* __restrict__ a, float* __restrict__ b, long n) {
  long i = (long)blockIdx.x * 256 + threadIdx.x;
  if (i < n) b[i] = a[i];
}
__global__ void concat_lat(const float* __restrict__ qa, const float* __restrict__ qt,
                           float* __restrict__ lat) {
  long i = (long)blockIdx.x * 256 + threadIdx.x;
  long tot = (long)Bz * 2 * NLt * Hd;
  if (i >= tot) return;
  int c = (int)(i % Hd);
  long t = i / Hd;
  int r = (int)(t % (2 * NLt));
  int bb = (int)(t / (2 * NLt));
  lat[i] = (r < NLt) ? qa[((long)bb * NLt + r) * Hd + c]
                     : qt[((long)bb * NLt + (r - NLt)) * Hd + c];
}

// mean over heads of probs: P[B,NH,Lq,Lk] -> out[B,Lq,Lk]
__global__ void mean_heads(const float* __restrict__ P, float* __restrict__ out,
                           int Lq, int Lk) {
  long i = (long)blockIdx.x * 256 + threadIdx.x;
  long tot = (long)Bz * Lq * Lk;
  if (i >= tot) return;
  int s = (int)(i % Lk);
  long t = i / Lk;
  int q = (int)(t % Lq);
  int bb = (int)(t / Lq);
  float acc = 0.f;
  for (int h = 0; h < NHd; ++h)
    acc += P[(((long)bb * NHd + h) * Lq + q) * Lk + s];
  out[i] = acc * (1.f / NHd);
}

// accumulate self-block probs into aas: acc[b,q,k] += sum_h P[b,h,q,NL+k]
__global__ void accum_probs(const float* __restrict__ P, float* __restrict__ acc) {
  int i = blockIdx.x * 256 + threadIdx.x;      // B*NL*NL
  if (i >= Bz * NLt * NLt) return;
  int k  = i % NLt;
  int q  = (i / NLt) % NLt;
  int bb = i / (NLt * NLt);
  float s = 0.f;
  for (int h = 0; h < NHd; ++h)
    s += P[(((long)bb * NHd + h) * (2 * NLt) + q) * (2 * NLt) + (NLt + k)];
  acc[i] += s;
}

// imp[b,s] = (1/(NL*L*NH)) * sum_k (sum_q aas[b,q,k]) * lat[b,k,s]
__global__ __launch_bounds__(1024) void importance(
    const float* __restrict__ aas, const float* __restrict__ lat,
    float* __restrict__ imp) {
  __shared__ float cs[NLt];
  int bb = blockIdx.x, t = threadIdx.x;        // blockDim = Sq = 1024
  if (t < NLt) {
    float s = 0.f;
    for (int q = 0; q < NLt; ++q) s += aas[((long)bb * NLt + q) * NLt + t];
    cs[t] = s;
  }
  __syncthreads();
  float v = 0.f;
  #pragma unroll
  for (int k = 0; k < NLt; ++k) v += cs[k] * lat[((long)bb * NLt + k) * Sq + t];
  imp[(long)bb * Sq + t] = v * (1.f / (NLt * Llay * NHd));
}

// per-batch top-K (stable first-occurrence tie-break), sorted indices,
// selection mask and rank (= # selected indices < s)
__global__ __launch_bounds__(256) void topk_sel(
    const float* __restrict__ imp, int* __restrict__ idx,
    int* __restrict__ sel, int* __restrict__ rank) {
  __shared__ float v[Sq];
  __shared__ int sidx[Ktop];
  int bb = blockIdx.x, t = threadIdx.x;
  for (int i = t; i < Sq; i += 256) v[i] = imp[(long)bb * Sq + i];
  __syncthreads();
  if (t == 0) {
    for (int j = 0; j < Ktop; ++j) {
      int best = 0; float bv = v[0];
      for (int i = 1; i < Sq; ++i) if (v[i] > bv) { bv = v[i]; best = i; }
      sidx[j] = best; v[best] = -3.4e38f;
    }
    for (int a = 1; a < Ktop; ++a) {           // sort ascending
      int key = sidx[a], p = a - 1;
      while (p >= 0 && sidx[p] > key) { sidx[p + 1] = sidx[p]; --p; }
      sidx[p + 1] = key;
    }
    for (int j = 0; j < Ktop; ++j) idx[bb * Ktop + j] = sidx[j];
  }
  __syncthreads();
  for (int s = t; s < Sq; s += 256) {
    int cnt = 0, is = 0;
    #pragma unroll
    for (int j = 0; j < Ktop; ++j) { cnt += (sidx[j] < s); is |= (sidx[j] == s); }
    sel[bb * Sq + s] = is;
    rank[bb * Sq + s] = cnt;
  }
}

// route audio rows: selected -> q_tokens[rank], unselected -> kv[s - rank]
__global__ __launch_bounds__(256) void gather_rows(
    const float* __restrict__ audio, const int* __restrict__ sel,
    const int* __restrict__ rank, float* __restrict__ qtok,
    float* __restrict__ kv) {
  int row = blockIdx.x;                         // Bz*Sq
  int bb = row / Sq, s = row % Sq;
  int is = sel[row], rk = rank[row];
  const float* src = audio + (long)row * Hd;
  float* dst = is ? (qtok + ((long)bb * Ktop + rk) * Hd)
                  : (kv + ((long)bb * Sq + (s - rk)) * Hd);
  for (int c = threadIdx.x; c < Hd; c += 256) dst[c] = src[c];
}

__global__ __launch_bounds__(256) void append_fused(
    const float* __restrict__ fused, float* __restrict__ kv) {
  int row = blockIdx.x;                         // Bz*2NL
  int bb = row / (2 * NLt), t = row % (2 * NLt);
  const float* src = fused + (long)row * Hd;
  float* dst = kv + ((long)bb * Sq + (Sq - Ktop) + t) * Hd;
  for (int c = threadIdx.x; c < Hd; c += 256) dst[c] = src[c];
}

// =====================================================================
// host-side orchestration
// =====================================================================
struct Scr { float *qb, *kb, *vb, *sb, *ob, *att; };

static inline void gemm(const float* A, const float* Bm, const float* bias, float* C,
                        int M, int N, int Kd, int lda, int ldb, int ldc,
                        int batches, int nh, long aO, long aI, long bO, long bI,
                        long cO, long cI, float alpha, bool transB, hipStream_t st) {
  dim3 grid((unsigned)((N + 127) / 128), (unsigned)((M + 127) / 128), (unsigned)batches);
  if (transB)
    wmma_gemm<true><<<grid, 256, 0, st>>>(A, Bm, bias, C, M, N, Kd, lda, ldb, ldc,
                                          nh, aO, aI, bO, bI, cO, cI, alpha);
  else
    wmma_gemm<false><<<grid, 256, 0, st>>>(A, Bm, bias, C, M, N, Kd, lda, ldb, ldc,
                                           nh, aO, aI, bO, bI, cO, cI, alpha);
}

static inline void ln(const float* x, const float* a, const float* g, const float* b,
                      float* out, int rows, hipStream_t st) {
  ln_residual<<<(rows + 7) / 8, 256, 0, st>>>(x, a, g, b, out, rows);
}

// one multi-head attention: out = softmax(QK^T/8 [+mask]) V @ Wo + bo
static void run_attn(const float* xq, const float* xkv,
                     const float* w, const float* bias,
                     int Lq, int Lk, const int* mask, int maskT,
                     float* out, float* probs_mean, float* aas_acc,
                     Scr& sc, hipStream_t st) {
  const size_t HH = (size_t)Hd * Hd;
  int Mq = Bz * Lq, Mk = Bz * Lk;
  gemm(xq,  w + 0 * HH, bias + 0 * Hd, sc.qb, Mq, Hd, Hd, Hd, Hd, Hd,
       1, 1, 0, 0, 0, 0, 0, 0, 1.f, false, st);
  gemm(xkv, w + 1 * HH, bias + 1 * Hd, sc.kb, Mk, Hd, Hd, Hd, Hd, Hd,
       1, 1, 0, 0, 0, 0, 0, 0, 1.f, false, st);
  gemm(xkv, w + 2 * HH, bias + 2 * Hd, sc.vb, Mk, Hd, Hd, Hd, Hd, Hd,
       1, 1, 0, 0, 0, 0, 0, 0, 1.f, false, st);
  // scores: per (b,h): [Lq x DH] x [Lk x DH]^T, scale 1/sqrt(64)
  gemm(sc.qb, sc.kb, nullptr, sc.sb, Lq, Lk, DHd, Hd, Hd, Lk,
       Bz * NHd, NHd, (long)Lq * Hd, DHd, (long)Lk * Hd, DHd,
       (long)NHd * Lq * Lk, (long)Lq * Lk, 0.125f, true, st);
  int rows = Bz * NHd * Lq;
  softmax_rows<<<(rows + 7) / 8, 256, 0, st>>>(sc.sb, mask, rows, Lk, NHd * Lq, maskT);
  if (probs_mean) {
    long tot = (long)Bz * Lq * Lk;
    mean_heads<<<(unsigned)((tot + 255) / 256), 256, 0, st>>>(sc.sb, probs_mean, Lq, Lk);
  }
  if (aas_acc)
    accum_probs<<<(Bz * NLt * NLt + 255) / 256, 256, 0, st>>>(sc.sb, aas_acc);
  // PV: per (b,h): [Lq x Lk] x [Lk x DH]
  gemm(sc.sb, sc.vb, nullptr, sc.ob, Lq, DHd, Lk, Lk, Hd, Hd,
       Bz * NHd, NHd, (long)NHd * Lq * Lk, (long)Lq * Lk, (long)Lk * Hd, DHd,
       (long)Lq * Hd, DHd, 1.f, false, st);
  gemm(sc.ob, w + 3 * HH, bias + 3 * Hd, out, Mq, Hd, Hd, Hd, Hd, Hd,
       1, 1, 0, 0, 0, 0, 0, 0, 1.f, false, st);
}

static void cross_block(float* q, int Lq, const float* enc, int Lk,
                        const float* w, const float* b, const float* g, const float* lb,
                        const int* mask, int maskT, float* probs_mean,
                        Scr& sc, hipStream_t st) {
  const size_t HH = (size_t)Hd * Hd;
  for (int l = 0; l < Llay; ++l) {
    const float* wl = w + (size_t)l * 8 * HH;
    const float* bl = b + (size_t)l * 8 * Hd;
    run_attn(q, q, wl, bl, Lq, Lq, nullptr, 0, sc.att, nullptr, nullptr, sc, st);
    ln(q, sc.att, g + ((size_t)l * 2 + 0) * Hd, lb + ((size_t)l * 2 + 0) * Hd, q, Bz * Lq, st);
    float* pm = (l == Llay - 1) ? probs_mean : nullptr;
    run_attn(q, enc, wl + 4 * HH, bl + 4 * Hd, Lq, Lk, mask, maskT, sc.att, pm, nullptr, sc, st);
    ln(q, sc.att, g + ((size_t)l * 2 + 1) * Hd, lb + ((size_t)l * 2 + 1) * Hd, q, Bz * Lq, st);
  }
}

static void self_block(float* q, const float* w, const float* b, const float* g,
                       const float* lb, float* aas, Scr& sc, hipStream_t st) {
  const size_t HH = (size_t)Hd * Hd;
  for (int l = 0; l < Llay; ++l) {
    run_attn(q, q, w + (size_t)l * 4 * HH, b + (size_t)l * 4 * Hd, 2 * NLt, 2 * NLt,
             nullptr, 0, sc.att, nullptr, aas, sc, st);
    ln(q, sc.att, g + (size_t)l * Hd, lb + (size_t)l * Hd, q, Bz * 2 * NLt, st);
  }
}

extern "C" void kernel_launch(void* const* d_in, const int* in_sizes, int n_in,
                              void* d_out, int out_size, void* d_ws, size_t ws_size,
                              hipStream_t stream) {
  (void)in_sizes; (void)n_in; (void)out_size; (void)ws_size;
  const float* audio    = (const float*)d_in[0];
  const float* text     = (const float*)d_in[1];
  const int*   amask    = (const int*)  d_in[2];
  const float* audio_qt = (const float*)d_in[3];
  const float* text_qt  = (const float*)d_in[4];
  const float* tm_w  = (const float*)d_in[5];
  const float* tm_b  = (const float*)d_in[6];
  const float* tm_g  = (const float*)d_in[7];
  const float* tm_lb = (const float*)d_in[8];
  const float* am_w  = (const float*)d_in[9];
  const float* am_b  = (const float*)d_in[10];
  const float* am_g  = (const float*)d_in[11];
  const float* am_lb = (const float*)d_in[12];
  const float* mg_w  = (const float*)d_in[13];
  const float* mg_b  = (const float*)d_in[14];
  const float* mg_g  = (const float*)d_in[15];
  const float* mg_lb = (const float*)d_in[16];
  const float* sel_w  = (const float*)d_in[17];
  const float* sel_b  = (const float*)d_in[18];
  const float* sel_g  = (const float*)d_in[19];
  const float* sel_lb = (const float*)d_in[20];

  // ---- workspace carve (floats, 16B-aligned chunks); ~445 MB total ----
  float* base = (float*)d_ws;
  size_t o = 0;
  auto grab = [&](size_t n) { float* p = base + o; o += (n + 3) & ~(size_t)3; return p; };
  Scr sc;
  sc.qb  = grab((size_t)Bz * 64 * Hd);
  sc.kb  = grab((size_t)Bz * Sq * Hd);
  sc.vb  = grab((size_t)Bz * Sq * Hd);
  sc.sb  = grab((size_t)Bz * NHd * 64 * Sq);
  sc.ob  = grab((size_t)Bz * 64 * Hd);
  sc.att = grab((size_t)Bz * 64 * Hd);
  float* qtext  = grab((size_t)Bz * NLt * Hd);
  float* qaudio = grab((size_t)Bz * NLt * Hd);
  float* latent = grab((size_t)Bz * 2 * NLt * Hd);
  float* lat_sc = grab((size_t)Bz * NLt * Sq);
  float* aas    = grab((size_t)Bz * NLt * NLt);
  float* imp    = grab((size_t)Bz * Sq);
  float* qtok   = grab((size_t)Bz * Ktop * Hd);
  float* kv     = grab((size_t)Bz * Sq * Hd);
  int* idx  = (int*)grab((size_t)Bz * Ktop);
  int* selm = (int*)grab((size_t)Bz * Sq);
  int* rank = (int*)grab((size_t)Bz * Sq);

  const long nqt = (long)Bz * NLt * Hd;

  // text token merger (masked cross-attn over text)
  bcast_qt<<<(unsigned)((nqt + 255) / 256), 256, 0, stream>>>(text_qt, qtext, nqt);
  cross_block(qtext, NLt, text, Tt, tm_w, tm_b, tm_g, tm_lb, amask, Tt, nullptr, sc, stream);

  // audio token merger (capture last-layer cross probs, head-mean)
  bcast_qt<<<(unsigned)((nqt + 255) / 256), 256, 0, stream>>>(audio_qt, qaudio, nqt);
  cross_block(qaudio, NLt, audio, Sq, am_w, am_b, am_g, am_lb, nullptr, 0, lat_sc, sc, stream);

  // latent = concat(audio, text); self block with prob accumulation
  const long nlat = (long)Bz * 2 * NLt * Hd;
  concat_lat<<<(unsigned)((nlat + 255) / 256), 256, 0, stream>>>(qaudio, qtext, latent);
  fill_zero<<<(Bz * NLt * NLt + 255) / 256, 256, 0, stream>>>(aas, (long)Bz * NLt * NLt);
  self_block(latent, sel_w, sel_b, sel_g, sel_lb, aas, sc, stream);

  // importance -> top-K -> token routing
  importance<<<Bz, Sq, 0, stream>>>(aas, lat_sc, imp);
  topk_sel<<<Bz, 256, 0, stream>>>(imp, idx, selm, rank);
  gather_rows<<<Bz * Sq, 256, 0, stream>>>(audio, selm, rank, qtok, kv);
  append_fused<<<Bz * 2 * NLt, 256, 0, stream>>>(latent, kv);

  // final merger cross block over kv = [rest ; fused]
  cross_block(qtok, Ktop, kv, Sq, mg_w, mg_b, mg_g, mg_lb, nullptr, 0, nullptr, sc, stream);

  const long nout = (long)Bz * Ktop * Hd;
  copyf<<<(unsigned)((nout + 255) / 256), 256, 0, stream>>>(qtok, (float*)d_out, nout);
}
